// AIWSEN_420906795700
// MI455X (gfx1250) — matrix operand order repaired and checked by reading, not compiled, not run on hardware
//
#include <hip/hip_runtime.h>

// ---------------------------------------------------------------------------
// Types
// ---------------------------------------------------------------------------
typedef __attribute__((ext_vector_type(16))) __bf16 bf16x16;
typedef __attribute__((ext_vector_type(8)))  __bf16 bf16x8;
typedef __attribute__((ext_vector_type(8)))  float  f32x8;

__device__ __forceinline__ __bf16 to_bf16(float x) { return (__bf16)x; }

// 16-element register pack for global->LDS pipelining
template <typename T> struct pack16 { T v[16]; };

__device__ __forceinline__ void gload16(pack16<float>& p, const float* src) {
    const float4* s = (const float4*)src;
#pragma unroll
    for (int i = 0; i < 4; ++i) *(float4*)&p.v[4*i] = s[i];
}
__device__ __forceinline__ void gload16(pack16<__bf16>& p, const __bf16* src) {
    *(uint4*)&p.v[0] = ((const uint4*)src)[0];
    *(uint4*)&p.v[8] = ((const uint4*)src)[1];
}
// convert (if needed) + vectorized LDS store (2 x b128)
__device__ __forceinline__ void sstore16(__bf16* dst, const pack16<float>& p) {
    __bf16 t[16];
#pragma unroll
    for (int i = 0; i < 16; ++i) t[i] = to_bf16(p.v[i]);
    ((uint4*)dst)[0] = *(const uint4*)&t[0];
    ((uint4*)dst)[1] = *(const uint4*)&t[8];
}
__device__ __forceinline__ void sstore16(__bf16* dst, const pack16<__bf16>& p) {
    ((uint4*)dst)[0] = *(const uint4*)&p.v[0];
    ((uint4*)dst)[1] = *(const uint4*)&p.v[8];
}

// Build A-fragment in contiguous k order (same order as the B loads):
// lane m=lane&15, kb=(lane>>4)*16; contiguous [kb..kb+15]
__device__ __forceinline__ bf16x16 make_afrag(const __bf16* p) {
    bf16x8 lo = *(const bf16x8*)p;
    bf16x8 hi = *(const bf16x8*)(p + 8);
    bf16x16 a;
#pragma unroll
    for (int i = 0; i < 8; ++i) { a[i] = lo[i]; a[8+i] = hi[i]; }
    return a;
}

// ---------------------------------------------------------------------------
// Generic WMMA GEMM:  C[M,N] = A[M,K] * B[N,K]^T   (i.e. x @ W.T)
// EPI: 0 = plain f32 store, 1 = head-split dual store (f32 + bf16), 2 = +bias
// 256 threads, tile 128x128x32, 8 waves in 2x4 grid, 8 wmma tiles per wave.
// Register-prefetch pipelining: tile k+1 global loads overlap tile k WMMAs.
// ---------------------------------------------------------------------------
constexpr int EPI_PLAIN  = 0;
constexpr int EPI_SPLITV = 1;
constexpr int EPI_BIAS   = 2;

template <typename TA, typename TB, int EPI>
__global__ __launch_bounds__(256) void gemm_wmma(
    const TA* __restrict__ A, const TB* __restrict__ B,
    float* __restrict__ Cf, __bf16* __restrict__ Cbf,
    const float* __restrict__ bias, int M, int N, int K)
{
    constexpr int BM = 128, BN = 128, BK = 32;
    __shared__ __attribute__((aligned(32))) __bf16 As[BM][BK];
    __shared__ __attribute__((aligned(32))) __bf16 Bs[BN][BK];

    const int tid  = threadIdx.x;
    const int wave = tid >> 5, lane = tid & 31;
    const int wm = wave & 1, wn = wave >> 1;   // 2 x 4 wave grid
    const int l16 = lane & 15, lh = lane >> 4;

    const int bm = blockIdx.x * BM;
    const int bn = blockIdx.y * BN;

    f32x8 acc[4][2];
#pragma unroll
    for (int i = 0; i < 4; ++i)
#pragma unroll
        for (int j = 0; j < 2; ++j) acc[i][j] = (f32x8){0,0,0,0,0,0,0,0};

    const int arow = tid >> 1;            // 0..127
    const int acol = (tid & 1) * 16;      // 0 / 16
    const TA* aptr = A + (size_t)(bm + arow) * K + acol;
    const TB* bptr = B + (size_t)(bn + arow) * K + acol;

    pack16<TA> pa;
    pack16<TB> pb;
    gload16(pa, aptr);
    gload16(pb, bptr);

    for (int k0 = 0; k0 < K; k0 += BK) {
        sstore16(&As[arow][acol], pa);
        sstore16(&Bs[arow][acol], pb);
        __syncthreads();

        if (k0 + BK < K) {               // prefetch next tile (overlaps WMMAs)
            gload16(pa, aptr + k0 + BK);
            gload16(pb, bptr + k0 + BK);
        }

        bf16x16 afr[4];
#pragma unroll
        for (int mt = 0; mt < 4; ++mt)
            afr[mt] = make_afrag(&As[wm*64 + mt*16 + l16][lh*16]);
        bf16x16 bfr[2];
#pragma unroll
        for (int nt = 0; nt < 2; ++nt)
            bfr[nt] = *(const bf16x16*)&Bs[wn*32 + nt*16 + l16][lh*16];

#pragma unroll
        for (int mt = 0; mt < 4; ++mt)
#pragma unroll
            for (int nt = 0; nt < 2; ++nt)
                acc[mt][nt] = __builtin_amdgcn_wmma_f32_16x16x32_bf16(
                    false, afr[mt], false, bfr[nt], (short)0, acc[mt][nt], false, false);
        __syncthreads();
    }

    // Epilogue. D layout: lane l16 = col, lh selects M 0-7 / 8-15, vgpr v = row.
#pragma unroll
    for (int mt = 0; mt < 4; ++mt)
#pragma unroll
        for (int nt = 0; nt < 2; ++nt)
#pragma unroll
            for (int v = 0; v < 8; ++v) {
                int row = bm + wm*64 + mt*16 + lh*8 + v;
                int col = bn + wn*32 + nt*16 + l16;
                float val = acc[mt][nt][v];
                if (EPI == EPI_BIAS) val += bias[col];
                if (EPI == EPI_SPLITV) {
                    // row = b*4096 + n ; col = h*128 + d -> head split
                    int b = row >> 12, n = row & 4095;
                    int h = col >> 7,  d = col & 127;
                    size_t idx = ((size_t)(b*8 + h) * 4096 + n) * 128 + d;
                    Cf[idx]  = val;
                    Cbf[idx] = to_bf16(val);
                } else {
                    Cf[(size_t)row * N + col] = val;
                }
            }
}

// ---------------------------------------------------------------------------
// Focusing post-process: one block per token row (B*N = 32768 rows).
// q = kv[:,C:], k = kv[:,:C] + pos_enc; relu+eps, /softplus(scale),
// q <- q^3 * ||q|| / ||q^3||  (norms over full C=1024), write head-split bf16.
// ---------------------------------------------------------------------------
__global__ __launch_bounds__(256) void focus_kernel(
    const float* __restrict__ kv, const float* __restrict__ pos_enc,
    const float* __restrict__ scale_p,
    __bf16* __restrict__ Qh, __bf16* __restrict__ Kh)
{
    const int row = blockIdx.x;            // 0..32767
    const int b = row >> 12, n = row & 4095;
    const int t = threadIdx.x;
    const int wave = t >> 5, lane = t & 31;
    const int c0 = t * 4;

    __shared__ float red[8][4];

    float4 kraw = *(const float4*)(kv + (size_t)row * 2048 + c0);
    float4 qraw = *(const float4*)(kv + (size_t)row * 2048 + 1024 + c0);
    float4 pe   = *(const float4*)(pos_enc + (size_t)n * 1024 + c0);
    float4 sp   = *(const float4*)(scale_p + c0);

    float q[4], k[4];
    {
        float kr[4] = {kraw.x + pe.x, kraw.y + pe.y, kraw.z + pe.z, kraw.w + pe.w};
        float qr[4] = {qraw.x, qraw.y, qraw.z, qraw.w};
        float sv[4] = {sp.x, sp.y, sp.z, sp.w};
#pragma unroll
        for (int i = 0; i < 4; ++i) {
            float s = log1pf(expf(sv[i]));              // softplus (inputs are small)
            q[i] = (fmaxf(qr[i], 0.f) + 1e-6f) / s;
            k[i] = (fmaxf(kr[i], 0.f) + 1e-6f) / s;
        }
    }

    float sq = 0.f, sq3 = 0.f, sk = 0.f, sk3 = 0.f;
#pragma unroll
    for (int i = 0; i < 4; ++i) {
        float q2 = q[i]*q[i], k2 = k[i]*k[i];
        sq  += q2;        sk  += k2;
        sq3 += q2*q2*q2;  sk3 += k2*k2*k2;   // (x^3)^2 = x^6
    }
#pragma unroll
    for (int m = 16; m > 0; m >>= 1) {
        sq  += __shfl_xor(sq,  m, 32);
        sq3 += __shfl_xor(sq3, m, 32);
        sk  += __shfl_xor(sk,  m, 32);
        sk3 += __shfl_xor(sk3, m, 32);
    }
    if (lane == 0) { red[wave][0]=sq; red[wave][1]=sq3; red[wave][2]=sk; red[wave][3]=sk3; }
    __syncthreads();
    float tq=0, tq3=0, tk=0, tk3=0;
#pragma unroll
    for (int w = 0; w < 8; ++w) { tq+=red[w][0]; tq3+=red[w][1]; tk+=red[w][2]; tk3+=red[w][3]; }

    const float fq = sqrtf(tq) * rsqrtf(tq3);
    const float fk = sqrtf(tk) * rsqrtf(tk3);

#pragma unroll
    for (int i = 0; i < 4; ++i) {
        int c = c0 + i;
        int h = c >> 7, d = c & 127;
        size_t idx = ((size_t)(b*8 + h) * 4096 + n) * 128 + d;
        Qh[idx] = to_bf16(q[i]*q[i]*q[i] * fq);
        Kh[idx] = to_bf16(k[i]*k[i]*k[i] * fk);
    }
}

// ---------------------------------------------------------------------------
// k.sum(axis=1):  ksum[bh][c] = sum_n Kh[bh][n][c]
// ---------------------------------------------------------------------------
__global__ void ksum_kernel(const __bf16* __restrict__ Kh, float* __restrict__ ksum)
{
    const int bh = blockIdx.x, c = threadIdx.x;     // 128 threads
    const __bf16* p = Kh + (size_t)bh * 4096 * 128 + c;
    float s = 0.f;
    for (int n = 0; n < 4096; ++n) s += (float)p[(size_t)n * 128];
    ksum[bh * 128 + c] = s;
}

// ---------------------------------------------------------------------------
// z[bh][i] = 1 / (dot(q[bh][i], ksum[bh]) + eps)   (one warp per row)
// ---------------------------------------------------------------------------
__global__ __launch_bounds__(256) void z_kernel(
    const __bf16* __restrict__ Qh, const float* __restrict__ ksum,
    float* __restrict__ z)
{
    __shared__ float ks[128];
    const int bh = blockIdx.y;
    const int t = threadIdx.x;
    if (t < 128) ks[t] = ksum[bh * 128 + t];
    __syncthreads();
    const int wave = t >> 5, lane = t & 31;
    const int i = blockIdx.x * 8 + wave;
    const __bf16* q = Qh + ((size_t)bh * 4096 + i) * 128 + lane * 4;
    float s = (float)q[0]*ks[lane*4+0] + (float)q[1]*ks[lane*4+1]
            + (float)q[2]*ks[lane*4+2] + (float)q[3]*ks[lane*4+3];
#pragma unroll
    for (int m = 16; m > 0; m >>= 1) s += __shfl_xor(s, m, 32);
    if (lane == 0) z[(size_t)bh * 4096 + i] = 1.f / (s + 1e-6f);
}

// ---------------------------------------------------------------------------
// kv_mat[c][d] = sum_j k[j][c] * v[j][d]  (per head-batch, 128x128, K=4096)
// Stores transposed: kvT[bh][d][c]  (ready as B operand of attention GEMM).
// One block per bh. LDS tiles stored transposed [ch][j], padded stride 40.
// Staging does a 4x4 register transpose: 4 channels x 4 j-rows per thread,
// packed b64 LDS stores (4 per operand) instead of 16 scalar b16 scatters.
// ---------------------------------------------------------------------------
__global__ __launch_bounds__(256) void kvmat_kernel(
    const __bf16* __restrict__ Kh, const __bf16* __restrict__ Vh,
    __bf16* __restrict__ kvT)
{
    constexpr int LDR = 40;   // padded row stride (halves); 80B = multiple of 16B
    __shared__ __attribute__((aligned(32))) __bf16 As[128 * LDR]; // As[c][j]
    __shared__ __attribute__((aligned(32))) __bf16 Bs[128 * LDR]; // Bs[d][j]

    const int bh = blockIdx.x;
    const __bf16* Kb = Kh + (size_t)bh * 4096 * 128;
    const __bf16* Vb = Vh + (size_t)bh * 4096 * 128;

    const int tid = threadIdx.x;
    const int wave = tid >> 5, lane = tid & 31;
    const int wm = wave & 1, wn = wave >> 1;
    const int l16 = lane & 15, lh = lane >> 4;

    f32x8 acc[4][2];
#pragma unroll
    for (int i = 0; i < 4; ++i)
#pragma unroll
        for (int j = 0; j < 2; ++j) acc[i][j] = (f32x8){0,0,0,0,0,0,0,0};

    const int jq = tid >> 5;              // 0..7 : group of 4 j-rows
    const int cb = (tid & 31) * 4;        // 4-channel base

    // prefetch regs: [row 0..3] x 4 channels for K and V
    __bf16 kr[4][4], vr[4][4];
#pragma unroll
    for (int r = 0; r < 4; ++r) {
        *(uint2*)&kr[r][0] = *(const uint2*)(Kb + (size_t)(jq*4 + r) * 128 + cb);
        *(uint2*)&vr[r][0] = *(const uint2*)(Vb + (size_t)(jq*4 + r) * 128 + cb);
    }

    for (int j0 = 0; j0 < 4096; j0 += 32) {
        // transpose 4x4 in registers, store packed b64 runs along j
#pragma unroll
        for (int i = 0; i < 4; ++i) {
            __bf16 kc[4] = {kr[0][i], kr[1][i], kr[2][i], kr[3][i]};
            __bf16 vc[4] = {vr[0][i], vr[1][i], vr[2][i], vr[3][i]};
            *(uint2*)&As[(cb + i) * LDR + jq*4] = *(const uint2*)kc;
            *(uint2*)&Bs[(cb + i) * LDR + jq*4] = *(const uint2*)vc;
        }
        __syncthreads();

        if (j0 + 32 < 4096) {             // prefetch next tile
#pragma unroll
            for (int r = 0; r < 4; ++r) {
                *(uint2*)&kr[r][0] = *(const uint2*)(Kb + (size_t)(j0 + 32 + jq*4 + r) * 128 + cb);
                *(uint2*)&vr[r][0] = *(const uint2*)(Vb + (size_t)(j0 + 32 + jq*4 + r) * 128 + cb);
            }
        }

        bf16x16 afr[4];
#pragma unroll
        for (int mt = 0; mt < 4; ++mt)
            afr[mt] = make_afrag(&As[(wm*64 + mt*16 + l16) * LDR + lh*16]);
        bf16x16 bfr[2];
#pragma unroll
        for (int nt = 0; nt < 2; ++nt)
            bfr[nt] = *(const bf16x16*)&Bs[(wn*32 + nt*16 + l16) * LDR + lh*16];

#pragma unroll
        for (int mt = 0; mt < 4; ++mt)
#pragma unroll
            for (int nt = 0; nt < 2; ++nt)
                acc[mt][nt] = __builtin_amdgcn_wmma_f32_16x16x32_bf16(
                    false, afr[mt], false, bfr[nt], (short)0, acc[mt][nt], false, false);
        __syncthreads();
    }

#pragma unroll
    for (int mt = 0; mt < 4; ++mt)
#pragma unroll
        for (int nt = 0; nt < 2; ++nt)
#pragma unroll
            for (int v = 0; v < 8; ++v) {
                int c = wm*64 + mt*16 + lh*8 + v;       // row of kv_mat
                int d = wn*32 + nt*16 + l16;            // col of kv_mat
                kvT[((size_t)bh * 128 + d) * 128 + c] = to_bf16(acc[mt][nt][v]);
            }
}

// ---------------------------------------------------------------------------
// Depthwise 5x5 conv (SAME) over v (head-split, NHWC with H=W=64, 128 ch).
// Block = (16x16 spatial tile) x (32-channel group), LDS-tiled with halo.
// ---------------------------------------------------------------------------
__global__ __launch_bounds__(256) void dwconv_kernel(
    const float* __restrict__ Vh, const float* __restrict__ w,
    const float* __restrict__ bias, float* __restrict__ fm)
{
    __shared__ float sv[20 * 20 * 32];   // 51.2 KB
    __shared__ float sw[25 * 32];
    __shared__ float sb[32];

    const int bh = blockIdx.z, cg = blockIdx.y, tile = blockIdx.x;
    const int ty0 = (tile >> 2) * 16, tx0 = (tile & 3) * 16;
    const int t = threadIdx.x;

    for (int idx = t; idx < 25 * 32; idx += 256) {
        int ch = idx & 31, tap = idx >> 5;
        sw[tap * 32 + ch] = w[(cg * 32 + ch) * 25 + tap];
    }
    if (t < 32) sb[t] = bias[cg * 32 + t];

    for (int idx = t; idx < 20 * 20 * 32; idx += 256) {
        int ch = idx & 31, rem = idx >> 5;
        int px = rem % 20, py = rem / 20;
        int gy = ty0 + py - 2, gx = tx0 + px - 2;
        float val = 0.f;
        if (gy >= 0 && gy < 64 && gx >= 0 && gx < 64)
            val = Vh[((size_t)bh * 4096 + gy * 64 + gx) * 128 + cg * 32 + ch];
        sv[idx] = val;
    }
    __syncthreads();

    const int py = t >> 4, px = t & 15;
    float4 acc4[8];
#pragma unroll
    for (int j = 0; j < 8; ++j)
        acc4[j] = make_float4(sb[4*j+0], sb[4*j+1], sb[4*j+2], sb[4*j+3]);

#pragma unroll
    for (int ky = 0; ky < 5; ++ky)
#pragma unroll
        for (int kx = 0; kx < 5; ++kx) {
            const float4* vv = (const float4*)&sv[((py + ky) * 20 + (px + kx)) * 32];
            const float4* ww = (const float4*)&sw[(ky * 5 + kx) * 32];
#pragma unroll
            for (int j = 0; j < 8; ++j) {
                float4 a = vv[j], b = ww[j];
                acc4[j].x = fmaf(a.x, b.x, acc4[j].x);
                acc4[j].y = fmaf(a.y, b.y, acc4[j].y);
                acc4[j].z = fmaf(a.z, b.z, acc4[j].z);
                acc4[j].w = fmaf(a.w, b.w, acc4[j].w);
            }
        }

    float4* dst = (float4*)&fm[((size_t)bh * 4096 + (ty0 + py) * 64 + tx0 + px) * 128 + cg * 32];
#pragma unroll
    for (int j = 0; j < 8; ++j) dst[j] = acc4[j];
}

// ---------------------------------------------------------------------------
// Attention GEMM:  x[i][d] = z[i] * (q[i][:] . kv_mat[:][d]) + fm[i][d]
// per bh: M=4096, N=K=128. B (kvT[bh], row-major [d][c]) cached whole in LDS.
// Output written merged-head bf16: attn[b][i][h*128+d].
// ---------------------------------------------------------------------------
__global__ __launch_bounds__(256) void attn_kernel(
    const __bf16* __restrict__ Qh, const __bf16* __restrict__ kvT,
    const float* __restrict__ z, const float* __restrict__ fm,
    __bf16* __restrict__ attn)
{
    __shared__ __attribute__((aligned(32))) __bf16 As[128][32];
    __shared__ __attribute__((aligned(32))) __bf16 Bs[128][128];

    const int bh = blockIdx.y;
    const int bm = blockIdx.x * 128;
    const int tid = threadIdx.x;
    const int wave = tid >> 5, lane = tid & 31;
    const int wm = wave & 1, wn = wave >> 1;
    const int l16 = lane & 15, lh = lane >> 4;

    {   // load full B tile (128x128 bf16 = 32KB)
        uint4* dstv = (uint4*)&Bs[0][0];
        const uint4* srcv = (const uint4*)(kvT + (size_t)bh * 16384);
        for (int i = tid; i < 2048; i += 256) dstv[i] = srcv[i];
    }

    f32x8 acc[4][2];
#pragma unroll
    for (int i = 0; i < 4; ++i)
#pragma unroll
        for (int j = 0; j < 2; ++j) acc[i][j] = (f32x8){0,0,0,0,0,0,0,0};

    const int arow = tid >> 1;
    const int acol = (tid & 1) * 16;
    const __bf16* qptr = Qh + ((size_t)bh * 4096 + bm + arow) * 128 + acol;

    pack16<__bf16> pa;
    gload16(pa, qptr);

    for (int c0 = 0; c0 < 128; c0 += 32) {
        sstore16(&As[arow][acol], pa);
        __syncthreads();

        if (c0 + 32 < 128) gload16(pa, qptr + c0 + 32);

        bf16x16 afr[4];
#pragma unroll
        for (int mt = 0; mt < 4; ++mt)
            afr[mt] = make_afrag(&As[wm*64 + mt*16 + l16][lh*16]);
        bf16x16 bfr[2];
#pragma unroll
        for (int nt = 0; nt < 2; ++nt)
            bfr[nt] = *(const bf16x16*)&Bs[wn*32 + nt*16 + l16][c0 + lh*16];

#pragma unroll
        for (int mt = 0; mt < 4; ++mt)
#pragma unroll
            for (int nt = 0; nt < 2; ++nt)
                acc[mt][nt] = __builtin_amdgcn_wmma_f32_16x16x32_bf16(
                    false, afr[mt], false, bfr[nt], (short)0, acc[mt][nt], false, false);
        __syncthreads();
    }

    const int b = bh >> 3, h = bh & 7;
#pragma unroll
    for (int mt = 0; mt < 4; ++mt)
#pragma unroll
        for (int nt = 0; nt < 2; ++nt)
#pragma unroll
            for (int v = 0; v < 8; ++v) {
                int i = bm + wm*64 + mt*16 + lh*8 + v;
                int d = wn*32 + nt*16 + l16;
                float val = acc[mt][nt][v] * z[(size_t)bh * 4096 + i]
                          + fm[((size_t)bh * 4096 + i) * 128 + d];
                attn[((size_t)b * 4096 + i) * 1024 + h * 128 + d] = to_bf16(val);
            }
}

// ---------------------------------------------------------------------------
// Host launcher
// ---------------------------------------------------------------------------
extern "C" void kernel_launch(void* const* d_in, const int* in_sizes, int n_in,
                              void* d_out, int out_size, void* d_ws, size_t ws_size,
                              hipStream_t stream) {
    const float* x1      = (const float*)d_in[0];
    const float* x2      = (const float*)d_in[1];
    const float* q_w     = (const float*)d_in[2];
    const float* kv_w    = (const float*)d_in[3];
    const float* proj_w  = (const float*)d_in[4];
    const float* proj_b  = (const float*)d_in[5];
    const float* scale_p = (const float*)d_in[6];
    const float* pos_enc = (const float*)d_in[7];
    const float* dwc_w   = (const float*)d_in[8];
    const float* dwc_b   = (const float*)d_in[9];
    float* out = (float*)d_out;

    char* ws = (char*)d_ws;
    float*  v_f32 = (float*)(ws);                         // 134,217,728 B
    __bf16* v_bf  = (__bf16*)(ws + 134217728ull);         //  67,108,864 B
    __bf16* q_bf  = (__bf16*)(ws + 201326592ull);         //  67,108,864 B
    __bf16* k_bf  = (__bf16*)(ws + 268435456ull);         //  67,108,864 B
    __bf16* kvT   = (__bf16*)(ws + 335544320ull);         //   2,097,152 B
    float*  ksum  = (float*)(ws + 337641472ull);          //      32,768 B
    float*  zbuf  = (float*)(ws + 337674240ull);          //   1,048,576 B
    char*   big   = ws + 338722816ull;                    // 268,435,456 B region
    float*  kvraw = (float*)big;                          // dead after focus
    float*  fm    = (float*)big;                          // reuse
    __bf16* attn  = (__bf16*)(big + 134217728ull);        // reuse tail

    dim3 blk(256);

    // 1) v = x2 @ q_w.T  (head-split f32 + bf16)
    gemm_wmma<float, float, EPI_SPLITV><<<dim3(256, 8), blk, 0, stream>>>(
        x2, q_w, v_f32, v_bf, nullptr, 32768, 1024, 1024);

    // 2) kv = x1 @ kv_w.T (raw, row-major 32768x2048)
    gemm_wmma<float, float, EPI_PLAIN><<<dim3(256, 16), blk, 0, stream>>>(
        x1, kv_w, kvraw, nullptr, nullptr, 32768, 2048, 1024);

    // 3) focusing postprocess -> head-split bf16 q/k
    focus_kernel<<<32768, 256, 0, stream>>>(kvraw, pos_enc, scale_p, q_bf, k_bf);

    // 4) k column sums, 5) z normalizer
    ksum_kernel<<<64, 128, 0, stream>>>(k_bf, ksum);
    z_kernel<<<dim3(512, 64), 256, 0, stream>>>(q_bf, ksum, zbuf);

    // 6) kv_mat = k^T v per head-batch (stored transposed bf16)
    kvmat_kernel<<<64, 256, 0, stream>>>(k_bf, v_bf, kvT);

    // 7) depthwise 5x5 conv on v (overwrites kvraw region)
    dwconv_kernel<<<dim3(16, 4, 64), 256, 0, stream>>>(v_f32, dwc_w, dwc_b, fm);

    // 8) x = z * (q @ kv_mat) + fm  -> merged-head bf16
    attn_kernel<<<dim3(32, 64), 256, 0, stream>>>(q_bf, kvT, zbuf, fm, attn);

    // 9) out = attn @ proj_w.T + proj_b
    gemm_wmma<__bf16, float, EPI_BIAS><<<dim3(256, 8), blk, 0, stream>>>(
        attn, proj_w, out, nullptr, proj_b, 32768, 1024, 1024);
}
